// AdS_GCL_73177652789489
// MI455X (gfx1250) — matrix operand order repaired and checked by reading, not compiled, not run on hardware
//
#include <hip/hip_runtime.h>
#include <hip/hip_bf16.h>

typedef __attribute__((ext_vector_type(16))) __bf16 v16bf;
typedef __attribute__((ext_vector_type(8)))  __bf16 v8bf;
typedef __attribute__((ext_vector_type(8)))  float  v8f;

#define FDIM 128
#define HDIM 128
#define EPW  64   // edges (or nodes) per workgroup
#define BLK  256  // 8 wave32

__device__ __forceinline__ unsigned short f2bf(float x) {
    unsigned int u = __float_as_uint(x);
    u += 0x7FFFu + ((u >> 16) & 1u);          // round-to-nearest-even
    return (unsigned short)(u >> 16);
}

__device__ __forceinline__ v16bf load_frag(const unsigned short* p0,
                                           const unsigned short* p1) {
    v8bf lo = *reinterpret_cast<const v8bf*>(p0);
    v8bf hi = *reinterpret_cast<const v8bf*>(p1);
    v16bf r;
#pragma unroll
    for (int i = 0; i < 8; ++i) { r[i] = lo[i]; r[i + 8] = hi[i]; }
    return r;
}

__device__ __forceinline__ float silu(float x) {
    return x / (1.f + __expf(-x));
}

// ---------------------------------------------------------------------------
// Repack f32 weight [K x 128] (row-major) into bf16 WMMA B-fragment order:
// dst[((nt*KT + kt)*32 + lane)*16 + j] = W[(kt*32 + (lane>>4)*8 + (j&7) + (j>>3)*16)*128 + nt*16 + (lane&15)]
// (B 32x16 bf16: lanes 0-15 hold K=0..7,16..23 of column lane, lanes 16-31 hold K=8..15,24..31)
// ---------------------------------------------------------------------------
__global__ void repack_w(const float* __restrict__ W, unsigned short* __restrict__ dst, int K) {
    int o = blockIdx.x * blockDim.x + threadIdx.x;
    if (o >= K * 128) return;
    int j    = o & 15;
    int lane = (o >> 4) & 31;
    int tile = o >> 9;
    int KT   = K >> 5;
    int kt   = tile % KT;
    int nt   = tile / KT;
    int k    = kt * 32 + (lane >> 4) * 8 + (j & 7) + (j >> 3) * 16;
    int c    = nt * 16 + (lane & 15);
    dst[o] = f2bf(W[k * 128 + c]);
}

// ---------------------------------------------------------------------------
// Edge kernel: fused dist + 2-layer edge MLP (WMMA) + atomic segment scatter
// ---------------------------------------------------------------------------
__global__ __launch_bounds__(BLK) void edge_kernel(
    const float* __restrict__ xz, const float* __restrict__ h,
    const float* __restrict__ We1,      // raw f32, for last row (dist column)
    const float* __restrict__ be1, const float* __restrict__ be2,
    const int* __restrict__ edge_index,
    const unsigned short* __restrict__ pW1,   // 256x128 frag-order bf16
    const unsigned short* __restrict__ pW2,   // 128x128 frag-order bf16
    float* __restrict__ num, float* __restrict__ deg,
    int E_, int N_)
{
    __shared__ __align__(16) unsigned short sA[EPW][2 * FDIM]; // [h_row | h_col] bf16
    __shared__ __align__(16) unsigned short sM[EPW][HDIM];     // layer-1 activations bf16
    __shared__ float sDist[EPW];
    __shared__ int   sRow[EPW];
    __shared__ int   sCol[EPW];

    const int t  = threadIdx.x;
    const int e0 = blockIdx.x * EPW;

    // ---- phase 0: edge geometry (AdS distance) + degree counts -----------
    if (t < EPW) {
        int  e  = e0 + t;
        bool ev = (e < E_);
        int  r  = ev ? edge_index[e]      : -1;
        int  c  = ev ? edge_index[E_ + e] : 0;
        sRow[t] = r;
        sCol[t] = c;
        float d = 0.f;
        if (ev) {
            float ax = xz[3 * r + 0], ay = xz[3 * r + 1], az = xz[3 * r + 2];
            float bx = xz[3 * c + 0], by = xz[3 * c + 1], bz = xz[3 * c + 2];
            float dx = ax - bx, dy = ay - by, dz = az - bz;
            float u  = (dx * dx + dy * dy + dz * dz) / (2.f * az * bz);
            float w  = 1.f + u;
            d = logf(w + sqrtf(w * w - 1.f));            // arccosh(1+u)
            __hip_atomic_fetch_add(&deg[r], 1.f, __ATOMIC_RELAXED,
                                   __HIP_MEMORY_SCOPE_AGENT);
        }
        sDist[t] = d;
    }
    __syncthreads();

    // ---- phase 1: gather h[row], h[col] -> LDS (bf16) ---------------------
    // 64 edges x 64 float4-chunks; chunk p<32 -> h[row], else h[col]
#pragma unroll
    for (int it = 0; it < (EPW * 64) / BLK; ++it) {
        int chunk = t + it * BLK;
        int e = chunk >> 6, p = chunk & 63;
        int node = (p < 32) ? max(sRow[e], 0) : sCol[e];
        int f4   = p & 31;
        float4 v = reinterpret_cast<const float4*>(h + (size_t)node * FDIM)[f4];
        ushort4 pk = { f2bf(v.x), f2bf(v.y), f2bf(v.z), f2bf(v.w) };
        *reinterpret_cast<ushort4*>(&sA[e][((p < 32) ? 0 : FDIM) + f4 * 4]) = pk;
    }
    __syncthreads();

    const int lane  = t & 31;
    const int wave  = t >> 5;
    const int mt    = wave & 3;            // 16-edge tile
    const int nbase = (wave >> 2) * 4;     // 4 column tiles of 16
    const int sub   = (lane >= 16) ? 8 : 0;
    const int cn    = lane & 15;
    const int cm    = (lane >> 4) * 8;

    // ---- phase 2: layer 1  (E x 256) @ (256 x 128) ------------------------
    v8f acc[4] = {};
    {
        const unsigned short* arow = &sA[mt * 16 + (lane & 15)][0];
#pragma unroll
        for (int kt = 0; kt < 8; ++kt) {
            v16bf a = load_frag(arow + kt * 32 + sub, arow + kt * 32 + 16 + sub);
#pragma unroll
            for (int n = 0; n < 4; ++n) {
                const unsigned short* bp = pW1 + (((nbase + n) * 8 + kt) * 32 + lane) * 16;
                v16bf b = load_frag(bp, bp + 8);
                acc[n] = __builtin_amdgcn_wmma_f32_16x16x32_bf16(
                    false, a, false, b, (short)0, acc[n], false, false);
            }
        }
    }
    // epilogue: + be1 + dist * We1[256,:], silu, -> sM (bf16)
#pragma unroll
    for (int n = 0; n < 4; ++n) {
        int col = (nbase + n) * 16 + cn;
        float bias  = be1[col];
        float wlast = We1[256 * 128 + col];
#pragma unroll
        for (int v = 0; v < 8; ++v) {
            int m = mt * 16 + cm + v;
            float x = acc[n][v] + bias + sDist[m] * wlast;
            sM[m][col] = f2bf(silu(x));
        }
    }
    __syncthreads();

    // ---- phase 3: layer 2  (E x 128) @ (128 x 128) ------------------------
    v8f acc2[4] = {};
    {
        const unsigned short* arow = &sM[mt * 16 + (lane & 15)][0];
#pragma unroll
        for (int kt = 0; kt < 4; ++kt) {
            v16bf a = load_frag(arow + kt * 32 + sub, arow + kt * 32 + 16 + sub);
#pragma unroll
            for (int n = 0; n < 4; ++n) {
                const unsigned short* bp = pW2 + (((nbase + n) * 4 + kt) * 32 + lane) * 16;
                v16bf b = load_frag(bp, bp + 8);
                acc2[n] = __builtin_amdgcn_wmma_f32_16x16x32_bf16(
                    false, a, false, b, (short)0, acc2[n], false, false);
            }
        }
    }
    // epilogue: + be2, silu, atomic scatter into num[row]
#pragma unroll
    for (int n = 0; n < 4; ++n) {
        int col = (nbase + n) * 16 + cn;
        float bias = be2[col];
#pragma unroll
        for (int v = 0; v < 8; ++v) {
            int m = mt * 16 + cm + v;
            int r = sRow[m];
            if (r >= 0) {
                float x = silu(acc2[n][v] + bias);
                __hip_atomic_fetch_add(&num[(size_t)r * HDIM + col], x,
                                       __ATOMIC_RELAXED, __HIP_MEMORY_SCOPE_AGENT);
            }
        }
    }
}

// ---------------------------------------------------------------------------
// Node kernel: agg = num/max(deg,1); out = h + silu([h|agg]@Wn1+bn1)@Wn2+bn2
// ---------------------------------------------------------------------------
__global__ __launch_bounds__(BLK) void node_kernel(
    const float* __restrict__ h,
    const float* __restrict__ bn1, const float* __restrict__ bn2,
    const unsigned short* __restrict__ pWn1,  // 256x128 frag-order
    const unsigned short* __restrict__ pWn2,  // 128x128 frag-order
    const float* __restrict__ num, const float* __restrict__ deg,
    float* __restrict__ out, int N_)
{
    __shared__ __align__(16) unsigned short sA[EPW][2 * FDIM];
    __shared__ __align__(16) unsigned short sZ[EPW][HDIM];

    const int t  = threadIdx.x;
    const int n0 = blockIdx.x * EPW;

    // stage [h | num/deg] as bf16
#pragma unroll
    for (int it = 0; it < (EPW * 64) / BLK; ++it) {
        int chunk = t + it * BLK;
        int nl = chunk >> 6, p = chunk & 63;
        int node = min(n0 + nl, N_ - 1);
        int f4 = p & 31;
        float4 v;
        if (p < 32) {
            v = reinterpret_cast<const float4*>(h + (size_t)node * FDIM)[f4];
        } else {
            v = reinterpret_cast<const float4*>(num + (size_t)node * HDIM)[f4];
            float inv = 1.f / fmaxf(deg[node], 1.f);
            v.x *= inv; v.y *= inv; v.z *= inv; v.w *= inv;
        }
        ushort4 pk = { f2bf(v.x), f2bf(v.y), f2bf(v.z), f2bf(v.w) };
        *reinterpret_cast<ushort4*>(&sA[nl][((p < 32) ? 0 : FDIM) + f4 * 4]) = pk;
    }
    __syncthreads();

    const int lane  = t & 31;
    const int wave  = t >> 5;
    const int mt    = wave & 3;
    const int nbase = (wave >> 2) * 4;
    const int sub   = (lane >= 16) ? 8 : 0;
    const int cn    = lane & 15;
    const int cm    = (lane >> 4) * 8;

    // layer 1: (N x 256) @ (256 x 128)
    v8f acc[4] = {};
    {
        const unsigned short* arow = &sA[mt * 16 + (lane & 15)][0];
#pragma unroll
        for (int kt = 0; kt < 8; ++kt) {
            v16bf a = load_frag(arow + kt * 32 + sub, arow + kt * 32 + 16 + sub);
#pragma unroll
            for (int n = 0; n < 4; ++n) {
                const unsigned short* bp = pWn1 + (((nbase + n) * 8 + kt) * 32 + lane) * 16;
                v16bf b = load_frag(bp, bp + 8);
                acc[n] = __builtin_amdgcn_wmma_f32_16x16x32_bf16(
                    false, a, false, b, (short)0, acc[n], false, false);
            }
        }
    }
#pragma unroll
    for (int n = 0; n < 4; ++n) {
        int col = (nbase + n) * 16 + cn;
        float bias = bn1[col];
#pragma unroll
        for (int v = 0; v < 8; ++v) {
            int m = mt * 16 + cm + v;
            sZ[m][col] = f2bf(silu(acc[n][v] + bias));
        }
    }
    __syncthreads();

    // layer 2: (N x 128) @ (128 x 128), + bn2 + residual
    v8f acc2[4] = {};
    {
        const unsigned short* arow = &sZ[mt * 16 + (lane & 15)][0];
#pragma unroll
        for (int kt = 0; kt < 4; ++kt) {
            v16bf a = load_frag(arow + kt * 32 + sub, arow + kt * 32 + 16 + sub);
#pragma unroll
            for (int n = 0; n < 4; ++n) {
                const unsigned short* bp = pWn2 + (((nbase + n) * 4 + kt) * 32 + lane) * 16;
                v16bf b = load_frag(bp, bp + 8);
                acc2[n] = __builtin_amdgcn_wmma_f32_16x16x32_bf16(
                    false, a, false, b, (short)0, acc2[n], false, false);
            }
        }
    }
#pragma unroll
    for (int n = 0; n < 4; ++n) {
        int col = (nbase + n) * 16 + cn;
        float bias = bn2[col];
#pragma unroll
        for (int v = 0; v < 8; ++v) {
            int m    = mt * 16 + cm + v;
            int node = n0 + m;
            if (node < N_) {
                size_t idx = (size_t)node * FDIM + col;
                out[idx] = h[idx] + acc2[n][v] + bias;
            }
        }
    }
}

extern "C" void kernel_launch(void* const* d_in, const int* in_sizes, int n_in,
                              void* d_out, int out_size, void* d_ws, size_t ws_size,
                              hipStream_t stream) {
    const float* xz  = (const float*)d_in[0];
    const float* h   = (const float*)d_in[1];
    const float* We1 = (const float*)d_in[2];   // (257,128)
    const float* be1 = (const float*)d_in[3];
    const float* We2 = (const float*)d_in[4];   // (128,128)
    const float* be2 = (const float*)d_in[5];
    const float* Wn1 = (const float*)d_in[6];   // (256,128)
    const float* bn1 = (const float*)d_in[7];
    const float* Wn2 = (const float*)d_in[8];   // (128,128)
    const float* bn2 = (const float*)d_in[9];
    const int* edge_index = (const int*)d_in[10];

    const int N = in_sizes[1] / FDIM;
    const int E = in_sizes[10] / 2;

    // workspace layout
    char* ws = (char*)d_ws;
    float* num = (float*)ws;                          size_t off = (size_t)N * HDIM * sizeof(float);
    float* deg = (float*)(ws + off);                  off += (size_t)N * sizeof(float);
    off = (off + 255) & ~(size_t)255;
    unsigned short* pW1  = (unsigned short*)(ws + off); off += 256 * 128 * sizeof(unsigned short);
    unsigned short* pW2  = (unsigned short*)(ws + off); off += 128 * 128 * sizeof(unsigned short);
    unsigned short* pWn1 = (unsigned short*)(ws + off); off += 256 * 128 * sizeof(unsigned short);
    unsigned short* pWn2 = (unsigned short*)(ws + off); off += 128 * 128 * sizeof(unsigned short);

    // zero segment-sum accumulators (num + deg contiguous)
    hipMemsetAsync(d_ws, 0, ((size_t)N * HDIM + N) * sizeof(float), stream);

    // repack weights into WMMA B-fragment order (bf16)
    repack_w<<<(256 * 128 + 255) / 256, 256, 0, stream>>>(We1, pW1, 256);
    repack_w<<<(128 * 128 + 255) / 256, 256, 0, stream>>>(We2, pW2, 128);
    repack_w<<<(256 * 128 + 255) / 256, 256, 0, stream>>>(Wn1, pWn1, 256);
    repack_w<<<(128 * 128 + 255) / 256, 256, 0, stream>>>(Wn2, pWn2, 128);

    // fused edge MLP + segment scatter
    edge_kernel<<<(E + EPW - 1) / EPW, BLK, 0, stream>>>(
        xz, h, We1, be1, be2, edge_index, pW1, pW2, num, deg, E, N);

    // fused node MLP + residual
    node_kernel<<<(N + EPW - 1) / EPW, BLK, 0, stream>>>(
        h, bn1, bn2, pWn1, pWn2, num, deg, (float*)d_out, N);
}
